// FSAPhonotacticsModel_24773371364107
// MI455X (gfx1250) — matrix operand order repaired and checked by reading, not compiled, not run on hardware
//
#include <hip/hip_runtime.h>
#include <hip/hip_bf16.h>

// CDNA5 / gfx1250: wave32, WMMA 16x16x32 f16 -> f32 accumulate.
typedef __attribute__((ext_vector_type(16))) _Float16 v16h;
typedef __attribute__((ext_vector_type(8)))  _Float16 v8h;
typedef __attribute__((ext_vector_type(8)))  float    v8f;

#if defined(__has_builtin)
#if __has_builtin(__builtin_amdgcn_sched_barrier)
#define SCHED_BARRIER() __builtin_amdgcn_sched_barrier(0)
#endif
#endif
#ifndef SCHED_BARRIER
#define SCHED_BARRIER()
#endif

constexpr int Q = 256;   // states
constexpr int S = 128;   // symbols
constexpr int B = 1024;  // batch
constexpr int T = 128;   // time

constexpr float STEP_SCALE = 1.0f / 256.0f;  // 2^-8 per-step rescale (growth ~= Q per step)
constexpr float LOG_SCALE_TOTAL = (float)T * 8.0f * 0.69314718055994530942f;

// -------- workspace layout (all offsets multiples of 256 bytes) --------
constexpr size_t OFF_PT    = 0;                                   // S*Q*Q f16 = 16 MiB
constexpr size_t OFF_PERM  = OFF_PT    + (size_t)S * Q * Q * 2;   // T*B i32  = 512 KiB
constexpr size_t OFF_START = OFF_PERM  + (size_t)T * B * 4;       // T*(S+1) i32
constexpr size_t OFF_A0    = OFF_START + (size_t)T * (S + 1) * 4; // B*Q f16
constexpr size_t OFF_A1    = OFF_A0    + (size_t)B * Q * 2;       // B*Q f16

// ---------------------------------------------------------------------
// 1) P^T[s][j][i] = exp(A[i][s][j]) as f16.  LDS-tiled transpose so both
//    the read of A (contiguous j) and write of P^T (contiguous i) coalesce.
__global__ void build_pt_kernel(const float* __restrict__ Alog,
                                _Float16* __restrict__ pt) {
  __shared__ float tile[16][17];
  const int s  = blockIdx.x;
  const int i0 = blockIdx.y * 16;
  const int j0 = blockIdx.z * 16;
  const int tx = threadIdx.x & 15;
  const int ty = threadIdx.x >> 4;
  const float v = Alog[((size_t)(i0 + ty) * S + s) * Q + (j0 + tx)];
  tile[ty][tx] = __expf(v);
  __syncthreads();
  // pt[s][j0+ty][i0+tx] = exp(A[i0+tx][s][j0+ty])
  pt[((size_t)s * Q + (j0 + ty)) * Q + (i0 + tx)] = (_Float16)tile[tx][ty];
}

// ---------------------------------------------------------------------
// 2) Per-timestep counting sort of batch elements by symbol.
//    starts[t][s] = exclusive prefix; perm[t][pos] = b.
__global__ void bucket_kernel(const int* __restrict__ xs,
                              int* __restrict__ perm,
                              int* __restrict__ starts) {
  __shared__ int cnt[S];
  __shared__ int cur[S];
  const int t = blockIdx.x;
  const int tid = threadIdx.x;
  if (tid < S) cnt[tid] = 0;
  __syncthreads();
  for (int b = tid; b < B; b += 256) atomicAdd(&cnt[xs[b * T + t]], 1);
  __syncthreads();
  if (tid == 0) {
    int acc = 0;
    for (int s = 0; s < S; ++s) {
      starts[t * (S + 1) + s] = acc;
      cur[s] = acc;
      acc += cnt[s];
    }
    starts[t * (S + 1) + S] = acc;
  }
  __syncthreads();
  for (int b = tid; b < B; b += 256) {
    const int s = xs[b * T + t];
    const int pos = atomicAdd(&cur[s], 1);
    perm[t * B + pos] = b;
  }
}

// ---------------------------------------------------------------------
// 3) alpha0[b][i] = linear-space init (reference: init>0 ? init : 0).
__global__ void init_alpha_kernel(const float* __restrict__ init,
                                  _Float16* __restrict__ a0) {
  const int b = blockIdx.x;
  const int j = threadIdx.x;
  const float v = init[j];
  a0[(size_t)b * Q + j] = (_Float16)(v > 0.0f ? v : 0.0f);
}

// ---------------------------------------------------------------------
// 4) One forward step: for each symbol bucket, alpha_new = (alpha @ P[s]) * 2^-8.
//    Block = one symbol (8 wave32s); wave w owns N-tiles [w*32, w*32+32).
//    B tiles (loop-invariant) live in 128 VGPRs for the whole block; the full
//    512B alpha row is loaded as ONE clause of 16x b128 loads, pinned above
//    the math with a sched_barrier so there is a single s_wait_loadcnt before
//    16 back-to-back v_wmma ops (instead of 8 exposed L2 round-trips).
//    WMMA operand layouts as loaded here (A and B share one K order):
//      A (16x32 f16): lane M = lane%16; elem e -> K = kb + (lane/16)*16 + e,
//                     (contiguous K per lane, same order as B).
//      B (32x16 f16): lane N = lane%16; elem e -> K = kb + (lane/16)*16 + e
//                     (contiguous K per lane -> read from transposed P).
//      D (16x16 f32): elem v -> M = v + (lane/16)*8, N = lane%16.
__global__ void __launch_bounds__(256) fsa_step_kernel(
    const _Float16* __restrict__ asrc, _Float16* __restrict__ adst,
    const _Float16* __restrict__ pt, const int* __restrict__ perm,
    const int* __restrict__ starts, int t) {
  const int s = blockIdx.x;
  const int start = starts[t * (S + 1) + s];
  const int count = starts[t * (S + 1) + s + 1] - start;
  if (count <= 0) return;  // block-uniform: EXEC stays all-ones for WMMA

  const int lane  = threadIdx.x & 31;
  const int wave  = threadIdx.x >> 5;   // 0..7
  const int j0    = wave * 32;          // two 16-wide N tiles per wave
  const int lidx  = lane & 15;
  const int khalf = lane >> 4;          // 0 or 1

  const int* pbase = perm + t * B + start;
  const _Float16* Bm  = pt + (size_t)s * Q * Q;
  const _Float16* br0 = Bm + (size_t)(j0 + lidx) * Q + khalf * 16;
  const _Float16* br1 = br0 + (size_t)16 * Q;

  // Preload the two B tiles for all 8 K-chunks (invariant over m0): 128 VGPRs.
  v16h b0[8], b1[8];
#pragma unroll
  for (int kk = 0; kk < 8; ++kk) {
    b0[kk] = *(const v16h*)(br0 + kk * 32);
    b1[kk] = *(const v16h*)(br1 + kk * 32);
  }

  for (int m0 = 0; m0 < count; m0 += 16) {
    const int r = m0 + lidx;
    const int bidx = pbase[(r < count) ? r : 0];  // dummy row for padding lanes
    const _Float16* arow = asrc + (size_t)bidx * Q + khalf * 16;

    // Entire alpha row as one clause of 16x b128 loads; keep all 64 VGPRs
    // live (do not let the scheduler re-roll into pairwise load+wait).
    v16h a[8];
#pragma unroll
    for (int kk = 0; kk < 8; ++kk) {
      const v8h alo = *(const v8h*)(arow + kk * 32);
      const v8h ahi = *(const v8h*)(arow + kk * 32 + 8);
#pragma unroll
      for (int e = 0; e < 8; ++e) { a[kk][e] = alo[e]; a[kk][8 + e] = ahi[e]; }
    }
    SCHED_BARRIER();  // pin all A loads above the WMMA chain

    v8f c0 = {};
    v8f c1 = {};
#pragma unroll
    for (int kk = 0; kk < 8; ++kk) {
      c0 = __builtin_amdgcn_wmma_f32_16x16x32_f16(false, a[kk], false, b0[kk],
                                                  (short)0, c0, false, false);
      c1 = __builtin_amdgcn_wmma_f32_16x16x32_f16(false, a[kk], false, b1[kk],
                                                  (short)0, c1, false, false);
    }
    SCHED_BARRIER();  // keep the store epilogue below the WMMA chain

    // Rescale, convert, scatter valid rows. Row->batch mapping for rows
    // m0..m0+15 already lives in lanes 0..15 (bidx): shuffle instead of
    // re-reading perm[] from memory.
#pragma unroll
    for (int v = 0; v < 8; ++v) {
      const int rowInTile = v + khalf * 8;          // 0..15
      const int bo = __shfl(bidx, rowInTile, 32);   // lane rowInTile holds it
      if (m0 + rowInTile < count) {
        adst[(size_t)bo * Q + (j0 + lidx)]      = (_Float16)(c0[v] * STEP_SCALE);
        adst[(size_t)bo * Q + (j0 + 16 + lidx)] = (_Float16)(c1[v] * STEP_SCALE);
      }
    }
  }
}

// ---------------------------------------------------------------------
// 5) LL[b] = log(sum_j alphaT[b][j] * exp(final[j])) + T*8*ln2.
__global__ void finalize_kernel(const _Float16* __restrict__ alphaT,
                                const float* __restrict__ fin,
                                float* __restrict__ out) {
  __shared__ float red[Q];
  const int b = blockIdx.x;
  const int j = threadIdx.x;
  red[j] = (float)alphaT[(size_t)b * Q + j] * __expf(fin[j]);
  __syncthreads();
  for (int off = Q / 2; off > 0; off >>= 1) {
    if (j < off) red[j] += red[j + off];
    __syncthreads();
  }
  if (j == 0) out[b] = logf(red[0]) + LOG_SCALE_TOTAL;
}

// ---------------------------------------------------------------------
extern "C" void kernel_launch(void* const* d_in, const int* in_sizes, int n_in,
                              void* d_out, int out_size, void* d_ws, size_t ws_size,
                              hipStream_t stream) {
  (void)in_sizes; (void)n_in; (void)out_size; (void)ws_size;
  const float* A    = (const float*)d_in[0];  // [Q,S,Q] log weights
  const float* init = (const float*)d_in[1];  // [Q] linear init
  const float* fin  = (const float*)d_in[2];  // [Q] log final
  const int*   xs   = (const int*)d_in[3];    // [B,T] symbols

  char* ws = (char*)d_ws;
  _Float16* pt   = (_Float16*)(ws + OFF_PT);
  int*      perm = (int*)(ws + OFF_PERM);
  int*      strt = (int*)(ws + OFF_START);
  _Float16* a0   = (_Float16*)(ws + OFF_A0);
  _Float16* a1   = (_Float16*)(ws + OFF_A1);

  // one-time-per-launch setup (still captured in the graph; all bandwidth-light)
  build_pt_kernel<<<dim3(S, Q / 16, Q / 16), 256, 0, stream>>>(A, pt);
  bucket_kernel<<<T, 256, 0, stream>>>(xs, perm, strt);
  init_alpha_kernel<<<B, Q, 0, stream>>>(init, a0);

  // T sequential scaled-forward steps, ping-ponging f16 alpha buffers
  for (int t = 0; t < T; ++t) {
    _Float16* src = (t & 1) ? a1 : a0;
    _Float16* dst = (t & 1) ? a0 : a1;
    fsa_step_kernel<<<S, 256, 0, stream>>>(src, dst, pt, perm, strt, t);
  }

  finalize_kernel<<<B, Q, 0, stream>>>((T & 1) ? a1 : a0, fin, (float*)d_out);
}
